// LocalCrossAttention_65300682769111
// MI455X (gfx1250) — compile-verified, matrix operand order checked by reading, not run on hardware
//
#include <hip/hip_runtime.h>
#include <hip/hip_bf16.h>

// ---------------------------------------------------------------- problem dims
constexpr int B_    = 4;
constexpr int CIN   = 256;
constexpr int DIMD  = 256;
constexpr int NQ_   = 2048;
constexpr int NK_   = 2048;
constexpr int KNN_  = 16;
constexpr int POSH  = 64;
constexpr int HID   = 1024;
constexpr int QT    = 4;      // queries per fused block -> 64 pixels = 4 WMMA col tiles

typedef __attribute__((ext_vector_type(16))) __bf16 v16bf;
typedef __attribute__((ext_vector_type(8)))  float  v8f;

// ---------------------------------------------------------------- workspace map
constexpr size_t OFF_Q    = 0;                                   // (B,DIM,NQ) f32
constexpr size_t OFF_KF   = OFF_Q    + (size_t)B_*DIMD*NQ_*4;    // (B,DIM,NK) f32
constexpr size_t OFF_V    = OFF_KF   + (size_t)B_*DIMD*NK_*4;    // (B,DIM,NK) f32
constexpr size_t OFF_QFB  = OFF_V    + (size_t)B_*DIMD*NK_*4;    // (B,CIN,NQ) bf16
constexpr size_t OFF_KFB  = OFF_QFB  + (size_t)B_*CIN*NQ_*2;     // (B,CIN,NK) bf16
constexpr size_t OFF_AGGB = OFF_KFB  + (size_t)B_*CIN*NK_*2;     // (B,DIM,NQ) bf16
constexpr size_t OFF_IDX  = OFF_AGGB + (size_t)B_*DIMD*NQ_*2;    // (B,NQ,K) i32
constexpr size_t OFF_WQS  = OFF_IDX  + (size_t)B_*NQ_*KNN_*4;    // swizzled bf16 weights
constexpr size_t OFF_WKS  = OFF_WQS  + (size_t)DIMD*CIN*2;
constexpr size_t OFF_WVS  = OFF_WKS  + (size_t)DIMD*CIN*2;
constexpr size_t OFF_WA1S = OFF_WVS  + (size_t)DIMD*CIN*2;
constexpr size_t OFF_WA2S = OFF_WA1S + (size_t)HID*DIMD*2;
constexpr size_t OFF_WES  = OFF_WA2S + (size_t)DIMD*HID*2;
constexpr size_t OFF_WP2S = OFF_WES  + (size_t)CIN*DIMD*2;
constexpr size_t OFF_S1   = OFF_WP2S + (size_t)DIMD*POSH*2;
constexpr size_t OFF_H1   = OFF_S1   + 256;
constexpr size_t OFF_S2   = OFF_H1   + 256;
constexpr size_t OFF_H2   = OFF_S2   + 4096;
// total ~39.9 MB

// ---------------------------------------------------------------- fragment swizzle
// CDNA5 16-bit A/B WMMA fragment layout (ISA 7.12.2). For K-index kk in [0,32)
// and row/col mn in [0,16): position inside the 512-bf16 fragment block.
__device__ __forceinline__ int frag_idx(int kk, int mn) {
  int hi    = (kk >> 3) & 1;                       // lanes 16-31 hold K 8-15 / 24-31
  int lane  = (hi << 4) + mn;
  int v     = ((kk & 16) >> 2) + ((kk & 7) >> 1);  // vgpr 0..7
  int which = kk & 1;                              // low/high half of dword
  return lane * 16 + v * 2 + which;
}

// ---------------------------------------------------------------- prep kernels
__global__ void cvt_f32_bf16(const float* __restrict__ s, __bf16* __restrict__ d, int n) {
  int i = blockIdx.x * blockDim.x + threadIdx.x;
  if (i < n) d[i] = (__bf16)s[i];
}

// Pre-swizzle a row-major fp32 weight (M x Kd) into A-fragment order (bf16).
__global__ void swizzleA(const float* __restrict__ W, __bf16* __restrict__ dst, int M, int Kd) {
  int total = M * Kd, nks = Kd >> 5;
  for (int e = blockIdx.x * blockDim.x + threadIdx.x; e < total; e += gridDim.x * blockDim.x) {
    int row = e / Kd, k = e - row * Kd;
    int mt = row >> 4, r = row & 15;
    int ks = k >> 5,  kk = k & 31;
    dst[((size_t)(mt * nks + ks) << 9) + frag_idx(kk, r)] = (__bf16)W[e];
  }
}

__global__ void fold_bn(const float* g, const float* be, const float* m, const float* v,
                        float* scale, float* shift, int n) {
  int i = blockIdx.x * blockDim.x + threadIdx.x;
  if (i < n) {
    float inv = g[i] * rsqrtf(v[i] + 1e-5f);
    scale[i] = inv;
    shift[i] = be[i] - m[i] * inv;
  }
}

// ---------------------------------------------------------------- KNN (top-16)
__global__ __launch_bounds__(128) void knn_kernel(const float* __restrict__ qpos,
                                                  const float* __restrict__ kpos,
                                                  int* __restrict__ idx) {
  __shared__ float kx[NK_], ky[NK_], kz[NK_];
  int blocksPerB = NQ_ / 128;
  int b  = blockIdx.x / blocksPerB;
  int q  = (blockIdx.x % blocksPerB) * 128 + threadIdx.x;
  const float* kp = kpos + (size_t)b * 3 * NK_;
  for (int i = threadIdx.x; i < NK_; i += 128) {
    kx[i] = kp[i]; ky[i] = kp[NK_ + i]; kz[i] = kp[2 * NK_ + i];
  }
  __syncthreads();
  const float* qp = qpos + (size_t)b * 3 * NQ_;
  float qx = qp[q], qy = qp[NQ_ + q], qz = qp[2 * NQ_ + q];
  float bd[KNN_]; int bi[KNN_];
#pragma unroll
  for (int j = 0; j < KNN_; ++j) { bd[j] = 3.4e38f; bi[j] = 0; }
  float worst = 3.4e38f; int wslot = 0;
  for (int nk = 0; nk < NK_; ++nk) {
    float dx = qx - kx[nk], dy = qy - ky[nk], dz = qz - kz[nk];
    float d = dx * dx + dy * dy + dz * dz;
    if (d < worst) {
      bd[wslot] = d; bi[wslot] = nk;
      worst = bd[0]; wslot = 0;
#pragma unroll
      for (int j = 1; j < KNN_; ++j) if (bd[j] > worst) { worst = bd[j]; wslot = j; }
    }
  }
  int* op = idx + ((size_t)b * NQ_ + q) * KNN_;
#pragma unroll
  for (int j = 0; j < KNN_; ++j) op[j] = bi[j];
}

// ---------------------------------------------------------------- WMMA GEMM (Kd=256)
// Out[b,row,n] = sum_k W[row,k]*X[b,k,n] + bias[row] (+ identity).
// Block: 256 thr (8 waves) -> 128(M) x 64(N) tile. Each wave owns one 16-row
// M-tile and reuses each A-fragment across 4 column tiles (weight traffic /4).
// All 4 B-fragments are loaded into registers before the WMMA burst so the
// ds_loads batch into one clause / one wait instead of load-wait-wmma x4.
__global__ __launch_bounds__(256) void gemm_bf16(const __bf16* __restrict__ Wsw,
                                                 const __bf16* __restrict__ Xb,
                                                 const float*  __restrict__ bias,
                                                 const float*  __restrict__ identity,
                                                 float* __restrict__ Out,
                                                 int M, int N) {
  __shared__ __bf16 xpanel[4 * 8 * 512];          // [ct][ks][frag] = 32 KB
  int ntiles = N >> 6, mblocks = M >> 7;
  int bid = blockIdx.x;
  int b   = bid / (ntiles * mblocks);
  int rem = bid % (ntiles * mblocks);
  int nt  = rem / mblocks;
  int mb  = rem % mblocks;
  int n0  = nt << 6;

  const __bf16* xsrc = Xb + (size_t)b * 256 * N;
  for (int e = threadIdx.x; e < 256 * 64; e += 256) {
    int k = e >> 6, c64 = e & 63;
    int ct = c64 >> 4, col = c64 & 15;
    xpanel[((ct * 8 + (k >> 5)) << 9) + frag_idx(k & 31, col)] = xsrc[(size_t)k * N + n0 + c64];
  }
  __syncthreads();

  int wave = threadIdx.x >> 5, lane = threadIdx.x & 31;
  int mt = (mb << 3) + wave;
  v8f acc[4];
#pragma unroll
  for (int ct = 0; ct < 4; ++ct) acc[ct] = (v8f){0.f,0.f,0.f,0.f,0.f,0.f,0.f,0.f};
  const __bf16* wp = Wsw + ((size_t)mt * 8 << 9) + lane * 16;
#pragma unroll
  for (int ks = 0; ks < 8; ++ks) {
    v16bf a = *(const v16bf*)(wp + (ks << 9));
    v16bf xb[4];
#pragma unroll
    for (int ct = 0; ct < 4; ++ct)
      xb[ct] = *(const v16bf*)(xpanel + ((ct * 8 + ks) << 9) + lane * 16);
#pragma unroll
    for (int ct = 0; ct < 4; ++ct)
      acc[ct] = __builtin_amdgcn_wmma_f32_16x16x32_bf16(false, a, false, xb[ct], (short)0, acc[ct], false, false);
  }
  int col = lane & 15, hi = lane >> 4;
#pragma unroll
  for (int ct = 0; ct < 4; ++ct) {
#pragma unroll
    for (int r = 0; r < 8; ++r) {
      int row = (mt << 4) + r + (hi << 3);
      size_t o = (size_t)b * M * N + (size_t)row * N + n0 + (ct << 4) + col;
      float vv = acc[ct][r] + bias[row];
      if (identity) vv += identity[o];
      Out[o] = vv;
    }
  }
}

// ---------------------------------------------------------------- fused attention
// One block = (batch b, 4 queries -> 64 pixels = 4 WMMA col tiles).
//   pos-rel -> wp1/BN/relu (VALU, tiny) -> pe1 bf16 B-fragments
//   PE = wp2 @ pe1 + bp2          (WMMA) ; X = qk_rel + PE -> bf16 B-fragments
//   A  = relu(BN(wa1 @ X + ba1))  (WMMA) -> bf16 B-fragments in LDS (128 KB)
//   L  = wa2 @ A + ba2            (WMMA) -> softmax over 16 neighbors (shfl_xor)
//   agg = sum_k att * (value_gathered + PE) -> bf16 (B,DIM,NQ)
// A-fragments loaded once per wave, reused across the 4 col tiles; B-fragments
// batch-loaded into registers ahead of each WMMA burst.
__global__ __launch_bounds__(256) void attn_fused(
    const float* __restrict__ qpos, const float* __restrict__ kpos,
    const int*   __restrict__ idx,
    const float* __restrict__ qproj, const float* __restrict__ kproj,
    const float* __restrict__ vproj,
    const float* __restrict__ wp1, const float* __restrict__ bp1,
    const float* __restrict__ s1,  const float* __restrict__ h1,
    const __bf16* __restrict__ wp2sw, const float* __restrict__ bp2,
    const __bf16* __restrict__ wa1sw, const float* __restrict__ ba1,
    const float* __restrict__ s2,  const float* __restrict__ h2,
    const __bf16* __restrict__ wa2sw, const float* __restrict__ ba2,
    __bf16* __restrict__ aggbf) {
  extern __shared__ __align__(32) char smem_raw[];
  float*  PE    = (float*)smem_raw;                         // 256*64 f32  (64 KB)
  __bf16* pe1sw = (__bf16*)(smem_raw + 65536);              // 4*2*512 bf16 (8 KB)
  __bf16* Xsw   = (__bf16*)(smem_raw + 65536 + 8192);       // 4*8*512 bf16 (32 KB)
  __bf16* Asw   = (__bf16*)(smem_raw + 65536 + 8192 + 32768); // 4*32*512 bf16 (128 KB)
  int*    sidx  = (int*)(smem_raw + 65536 + 8192 + 32768 + 131072); // 64 ints

  int blocksPerB = NQ_ / QT;
  int b  = blockIdx.x / blocksPerB;
  int n0 = (blockIdx.x % blocksPerB) * QT;

  if (threadIdx.x < QT * KNN_) {
    int p = threadIdx.x;
    sidx[p] = idx[((size_t)b * NQ_ + n0 + (p >> 4)) * KNN_ + (p & 15)];
  }
  __syncthreads();

  // --- stage 0a: pe1 = relu(BN(wp1 @ pos_rel)) -> bf16 B-fragments
  for (int e = threadIdx.x; e < POSH * 64; e += 256) {
    int h = e >> 6, p = e & 63;
    int q = n0 + (p >> 4), nk = sidx[p];
    const float* qp = qpos + (size_t)b * 3 * NQ_;
    const float* kp = kpos + (size_t)b * 3 * NK_;
    float px = qp[q]           - kp[nk];
    float py = qp[NQ_ + q]     - kp[NK_ + nk];
    float pz = qp[2 * NQ_ + q] - kp[2 * NK_ + nk];
    float t = wp1[h * 3] * px + wp1[h * 3 + 1] * py + wp1[h * 3 + 2] * pz + bp1[h];
    t = fmaxf(t * s1[h] + h1[h], 0.f);
    pe1sw[(((p >> 4) * 2 + (h >> 5)) << 9) + frag_idx(h & 31, p & 15)] = (__bf16)t;
  }
  __syncthreads();

  int wave = threadIdx.x >> 5, lane = threadIdx.x & 31;
  int col = lane & 15, hi = lane >> 4;

  // --- stage 0b: PE = wp2 @ pe1 + bp2 (WMMA); X = qk_rel + PE -> bf16 frags
  for (int mt = wave; mt < DIMD / 16; mt += 8) {
    v8f acc[QT];
#pragma unroll
    for (int ct = 0; ct < QT; ++ct) acc[ct] = (v8f){0.f,0.f,0.f,0.f,0.f,0.f,0.f,0.f};
    const __bf16* wp = wp2sw + ((size_t)mt * 2 << 9) + lane * 16;
#pragma unroll
    for (int ks = 0; ks < 2; ++ks) {
      v16bf a = *(const v16bf*)(wp + (ks << 9));
      v16bf pb[QT];
#pragma unroll
      for (int ct = 0; ct < QT; ++ct)
        pb[ct] = *(const v16bf*)(pe1sw + ((ct * 2 + ks) << 9) + lane * 16);
#pragma unroll
      for (int ct = 0; ct < QT; ++ct)
        acc[ct] = __builtin_amdgcn_wmma_f32_16x16x32_bf16(false, a, false, pb[ct], (short)0, acc[ct], false, false);
    }
#pragma unroll
    for (int ct = 0; ct < QT; ++ct) {
      int p = (ct << 4) + col;
      int q = n0 + ct, nk = sidx[p];
#pragma unroll
      for (int r = 0; r < 8; ++r) {
        int c = (mt << 4) + r + (hi << 3);
        float pe = acc[ct][r] + bp2[c];
        PE[(c << 6) + p] = pe;
        float xv = qproj[((size_t)b * DIMD + c) * NQ_ + q]
                 - kproj[((size_t)b * DIMD + c) * NK_ + nk] + pe;
        Xsw[((ct * 8 + (c >> 5)) << 9) + frag_idx(c & 31, col)] = (__bf16)xv;
      }
    }
  }
  __syncthreads();

  // --- stage 1: A = relu(BN(wa1 @ X + ba1)); A-fragment reused over 4 col tiles
  for (int mt = wave; mt < HID / 16; mt += 8) {
    v8f acc[QT];
#pragma unroll
    for (int ct = 0; ct < QT; ++ct) acc[ct] = (v8f){0.f,0.f,0.f,0.f,0.f,0.f,0.f,0.f};
    const __bf16* wp = wa1sw + ((size_t)mt * 8 << 9) + lane * 16;
    if (mt + 8 < HID / 16)
      __builtin_prefetch((const void*)(wp + ((size_t)8 * 8 << 9)), 0, 1);
#pragma unroll
    for (int ks = 0; ks < 8; ++ks) {
      v16bf a = *(const v16bf*)(wp + (ks << 9));
      v16bf xb[QT];
#pragma unroll
      for (int ct = 0; ct < QT; ++ct)
        xb[ct] = *(const v16bf*)(Xsw + ((ct * 8 + ks) << 9) + lane * 16);
#pragma unroll
      for (int ct = 0; ct < QT; ++ct)
        acc[ct] = __builtin_amdgcn_wmma_f32_16x16x32_bf16(false, a, false, xb[ct], (short)0, acc[ct], false, false);
    }
#pragma unroll
    for (int ct = 0; ct < QT; ++ct) {
#pragma unroll
      for (int r = 0; r < 8; ++r) {
        int row = (mt << 4) + r + (hi << 3);        // hidden channel 0..1023
        float av = fmaxf((acc[ct][r] + ba1[row]) * s2[row] + h2[row], 0.f);
        Asw[((ct * 32 + (row >> 5)) << 9) + frag_idx(row & 31, col)] = (__bf16)av;
      }
    }
  }
  __syncthreads();

  // --- stage 2: logits = wa2 @ A + ba2; softmax over 16 neighbors; aggregate
  for (int mt = wave; mt < DIMD / 16; mt += 8) {
    v8f acc[QT];
#pragma unroll
    for (int ct = 0; ct < QT; ++ct) acc[ct] = (v8f){0.f,0.f,0.f,0.f,0.f,0.f,0.f,0.f};
    const __bf16* wp = wa2sw + ((size_t)mt * 32 << 9) + lane * 16;
    if (mt + 8 < DIMD / 16)
      __builtin_prefetch((const void*)(wp + ((size_t)8 * 32 << 9)), 0, 1);
#pragma unroll
    for (int ks = 0; ks < 32; ++ks) {
      v16bf a = *(const v16bf*)(wp + (ks << 9));
      v16bf ab[QT];
#pragma unroll
      for (int ct = 0; ct < QT; ++ct)
        ab[ct] = *(const v16bf*)(Asw + ((ct * 32 + ks) << 9) + lane * 16);
#pragma unroll
      for (int ct = 0; ct < QT; ++ct)
        acc[ct] = __builtin_amdgcn_wmma_f32_16x16x32_bf16(false, a, false, ab[ct], (short)0, acc[ct], false, false);
    }
#pragma unroll
    for (int ct = 0; ct < QT; ++ct) {
      int p = (ct << 4) + col;
      int nk = sidx[p];
#pragma unroll
      for (int r = 0; r < 8; ++r) {
        int row = (mt << 4) + r + (hi << 3);        // output channel 0..255
        float lg = acc[ct][r] + ba2[row];
        float mx = lg;                               // max over the 16 cols
#pragma unroll
        for (int off = 1; off < 16; off <<= 1) mx = fmaxf(mx, __shfl_xor(mx, off, 32));
        float ex = __expf(lg - mx);
        float sm = ex;
#pragma unroll
        for (int off = 1; off < 16; off <<= 1) sm += __shfl_xor(sm, off, 32);
        float att = ex / sm;
        float vwp = vproj[((size_t)b * DIMD + row) * NK_ + nk] + PE[(row << 6) + p];
        float contrib = att * vwp;
#pragma unroll
        for (int off = 1; off < 16; off <<= 1) contrib += __shfl_xor(contrib, off, 32);
        if (col == 0)
          aggbf[((size_t)b * DIMD + row) * NQ_ + n0 + ct] = (__bf16)contrib;
      }
    }
  }
}

// ---------------------------------------------------------------- host launcher
extern "C" void kernel_launch(void* const* d_in, const int* in_sizes, int n_in,
                              void* d_out, int out_size, void* d_ws, size_t ws_size,
                              hipStream_t stream) {
  const float* query_pos  = (const float*)d_in[0];
  const float* query_feat = (const float*)d_in[1];
  const float* key_pos    = (const float*)d_in[2];
  const float* key_feat   = (const float*)d_in[3];
  const float* wq  = (const float*)d_in[4];  const float* bq  = (const float*)d_in[5];
  const float* wk  = (const float*)d_in[6];  const float* bk  = (const float*)d_in[7];
  const float* wv  = (const float*)d_in[8];  const float* bv  = (const float*)d_in[9];
  const float* wp1 = (const float*)d_in[10]; const float* bp1 = (const float*)d_in[11];
  const float* g1  = (const float*)d_in[12]; const float* be1 = (const float*)d_in[13];
  const float* m1  = (const float*)d_in[14]; const float* v1  = (const float*)d_in[15];
  const float* wp2 = (const float*)d_in[16]; const float* bp2 = (const float*)d_in[17];
  const float* wa1 = (const float*)d_in[18]; const float* ba1 = (const float*)d_in[19];
  const float* g2  = (const float*)d_in[20]; const float* be2 = (const float*)d_in[21];
  const float* m2  = (const float*)d_in[22]; const float* v2  = (const float*)d_in[23];
  const float* wa2 = (const float*)d_in[24]; const float* ba2 = (const float*)d_in[25];
  const float* we  = (const float*)d_in[26]; const float* bE  = (const float*)d_in[27];

  char* ws = (char*)d_ws;
  float*  q_f32  = (float*)(ws + OFF_Q);
  float*  kf_f32 = (float*)(ws + OFF_KF);
  float*  v_f32  = (float*)(ws + OFF_V);
  __bf16* qfb    = (__bf16*)(ws + OFF_QFB);
  __bf16* kfb    = (__bf16*)(ws + OFF_KFB);
  __bf16* aggbf  = (__bf16*)(ws + OFF_AGGB);
  int*    idxb   = (int*)(ws + OFF_IDX);
  __bf16* wqs    = (__bf16*)(ws + OFF_WQS);
  __bf16* wks    = (__bf16*)(ws + OFF_WKS);
  __bf16* wvs    = (__bf16*)(ws + OFF_WVS);
  __bf16* wa1s   = (__bf16*)(ws + OFF_WA1S);
  __bf16* wa2s   = (__bf16*)(ws + OFF_WA2S);
  __bf16* wes    = (__bf16*)(ws + OFF_WES);
  __bf16* wp2s   = (__bf16*)(ws + OFF_WP2S);
  float*  s1     = (float*)(ws + OFF_S1);
  float*  h1     = (float*)(ws + OFF_H1);
  float*  s2     = (float*)(ws + OFF_S2);
  float*  h2     = (float*)(ws + OFF_H2);

  // 1) bf16 feature copies + weight swizzles + BN folding
  int nfeat = B_ * CIN * NQ_;
  cvt_f32_bf16<<<(nfeat + 255) / 256, 256, 0, stream>>>(query_feat, qfb, nfeat);
  cvt_f32_bf16<<<(nfeat + 255) / 256, 256, 0, stream>>>(key_feat,   kfb, nfeat);
  swizzleA<<<256, 256, 0, stream>>>(wq,  wqs,  DIMD, CIN);
  swizzleA<<<256, 256, 0, stream>>>(wk,  wks,  DIMD, CIN);
  swizzleA<<<256, 256, 0, stream>>>(wv,  wvs,  DIMD, CIN);
  swizzleA<<<256, 256, 0, stream>>>(wa1, wa1s, HID,  DIMD);
  swizzleA<<<256, 256, 0, stream>>>(wa2, wa2s, DIMD, HID);
  swizzleA<<<256, 256, 0, stream>>>(we,  wes,  CIN,  DIMD);
  swizzleA<<<64,  256, 0, stream>>>(wp2, wp2s, DIMD, POSH);
  fold_bn<<<1, POSH, 0, stream>>>(g1, be1, m1, v1, s1, h1, POSH);
  fold_bn<<<(HID + 255) / 256, 256, 0, stream>>>(g2, be2, m2, v2, s2, h2, HID);

  // 2) KNN indices
  knn_kernel<<<B_ * (NQ_ / 128), 128, 0, stream>>>(query_pos, key_pos, idxb);

  // 3) q/k/v projections (WMMA)
  int gemm_grid = B_ * (NQ_ / 64) * (DIMD / 128);
  gemm_bf16<<<gemm_grid, 256, 0, stream>>>(wqs, qfb, bq, nullptr, q_f32,  DIMD, NQ_);
  gemm_bf16<<<gemm_grid, 256, 0, stream>>>(wks, kfb, bk, nullptr, kf_f32, DIMD, NK_);
  gemm_bf16<<<gemm_grid, 256, 0, stream>>>(wvs, kfb, bv, nullptr, v_f32,  DIMD, NK_);

  // 4) fused local attention (PE-WMMA + wa1-WMMA + wa2-WMMA + softmax + agg)
  size_t smem_attn = 65536            // PE f32
                   + 8192             // pe1 bf16 frags
                   + 32768            // X bf16 frags
                   + 131072           // A bf16 frags
                   + QT * KNN_ * 4;   // neighbor indices
  attn_fused<<<B_ * (NQ_ / QT), 256, smem_attn, stream>>>(
      query_pos, key_pos, idxb, q_f32, kf_f32, v_f32,
      wp1, bp1, s1, h1, wp2s, bp2,
      wa1s, ba1, s2, h2, wa2s, ba2, aggbf);

  // 5) final projection + residual
  gemm_bf16<<<gemm_grid, 256, 0, stream>>>(wes, aggbf, bE, query_feat,
                                           (float*)d_out, CIN, NQ_);
}
